// MotionNet_Decoder_71889162600859
// MI455X (gfx1250) — matrix operand order repaired and checked by reading, not compiled, not run on hardware
//
#include <hip/hip_runtime.h>
#include <stdint.h>

// ---------- types ----------
typedef __bf16        bf16x16 __attribute__((ext_vector_type(16)));
typedef float         f32x8   __attribute__((ext_vector_type(8)));
typedef unsigned int  u32x8   __attribute__((ext_vector_type(8)));
typedef int           v4i     __attribute__((vector_size(16)));   // matches builtin param
typedef __attribute__((address_space(1))) v4i* gv4i_p;            // global int4*
typedef __attribute__((address_space(3))) v4i* lv4i_p;            // LDS int4*

// ---------- async global->LDS (CDNA5), with fallback ----------
#if defined(__has_builtin)
#if __has_builtin(__builtin_amdgcn_global_load_async_to_lds_b128) && \
    __has_builtin(__builtin_amdgcn_s_wait_asynccnt)
#define USE_ASYNC_LDS 1
#endif
#endif

__device__ __forceinline__ void stage_b128(const unsigned int* gsrc, unsigned int* ldst) {
#if defined(USE_ASYNC_LDS)
  __builtin_amdgcn_global_load_async_to_lds_b128(
      (gv4i_p)(void*)gsrc, (lv4i_p)(void*)ldst, 0, 0);
#else
  uint4 v = *(const uint4*)gsrc;
  *(uint4*)ldst = v;
#endif
}
__device__ __forceinline__ void wait_async1() {
#if defined(USE_ASYNC_LDS)
  __builtin_amdgcn_s_wait_asynccnt(1);
#endif
}
__device__ __forceinline__ void wait_async0() {
#if defined(USE_ASYNC_LDS)
  __builtin_amdgcn_s_wait_asynccnt(0);
#endif
}

// ---------- helpers ----------
__device__ __forceinline__ unsigned short f2bf(float f) {
  unsigned int u = __float_as_uint(f);
  u += 0x7fffu + ((u >> 16) & 1u);          // round-to-nearest-even
  return (unsigned short)(u >> 16);
}
__device__ __forceinline__ unsigned int pack2(float lo, float hi) {
  return (unsigned int)f2bf(lo) | ((unsigned int)f2bf(hi) << 16);
}
__device__ __forceinline__ float elu1(float x) { return x > 0.f ? x : (__expf(x) - 1.f); }
__device__ __forceinline__ void stout(float* p, float v) { *p = v; }
__device__ __forceinline__ void stout(unsigned short* p, float v) { *p = f2bf(v); }
__device__ __forceinline__ f32x8 zero8() {
  f32x8 v;
#pragma unroll
  for (int i = 0; i < 8; ++i) v[i] = 0.f;
  return v;
}

// A fragment (16x32 bf16, wave32 layout): row m = lane&15, hl = lane>>4.
// VGPR0..3 = K {0..7}+8*hl packed pairs, VGPR4..7 = K {16..23}+8*hl.
__device__ __forceinline__ u32x8 load_a_frag(const unsigned short* arow, int k0, int hl) {
  u32x8 a;
  uint4 x = *(const uint4*)(arow + k0 + hl * 8);
  uint4 y = *(const uint4*)(arow + k0 + 16 + hl * 8);
  a[0] = x.x; a[1] = x.y; a[2] = x.z; a[3] = x.w;
  a[4] = y.x; a[5] = y.y; a[6] = y.z; a[7] = y.w;
  return a;
}
__device__ __forceinline__ u32x8 load_a_frag(const float* arow, int k0, int hl) {
  u32x8 a;
  const float4* p0 = (const float4*)(arow + k0 + hl * 8);
  const float4* p1 = (const float4*)(arow + k0 + 16 + hl * 8);
  float4 x0 = p0[0], x1 = p0[1], y0 = p1[0], y1 = p1[1];
  a[0] = pack2(x0.x, x0.y); a[1] = pack2(x0.z, x0.w);
  a[2] = pack2(x1.x, x1.y); a[3] = pack2(x1.z, x1.w);
  a[4] = pack2(y0.x, y0.y); a[5] = pack2(y0.z, y0.w);
  a[6] = pack2(y1.x, y1.y); a[7] = pack2(y1.z, y1.w);
  return a;
}

// one K-step of MMA: 4 subtile fragments from LDS, 4 WMMAs against shared A
__device__ __forceinline__ void mma_step(const unsigned int* lb, int lane,
                                         u32x8 au, f32x8* acc) {
  bf16x16 av = __builtin_bit_cast(bf16x16, au);
#pragma unroll
  for (int j = 0; j < 4; ++j) {
    int base = j * 384 + lane * 12;
    uint4 b0 = *(const uint4*)&lb[base];        // ds_load_b128
    uint4 b1 = *(const uint4*)&lb[base + 4];
    u32x8 bu;
    bu[0] = b0.x; bu[1] = b0.y; bu[2] = b0.z; bu[3] = b0.w;
    bu[4] = b1.x; bu[5] = b1.y; bu[6] = b1.z; bu[7] = b1.w;
    bf16x16 bv = __builtin_bit_cast(bf16x16, bu);
    acc[j] = __builtin_amdgcn_wmma_f32_16x16x32_bf16(
        false, av, false, bv, (short)0, acc[j], false, false);
  }
}

// ---------- fused (MoE-)GEMM: C = act( sum_e omega_e * (A @ W_e + b_e) ) ----------
// A : [16384, lda] bf16(u16) or fp32; K-loop bound kp (mult of 32, <= lda)
// Wp: fragment-swizzled bf16-pair dwords, [EXP][kp/32][npad/16][lane=32][r=8]
// bias: fp32 [EXP][npad]; omega: fp32 [16384,8] (EXP>1 only)
// Block = 256 thr = 8 waves; block tile 128M x 64N; wave tile 16M x 64N.
// B tile double-buffered in LDS, staged with async b128 copies (1 per thread);
// last K-step peeled so the steady-state loop is branch-free.
template <int EXP, bool ACT, typename AT, typename OT>
__global__ __launch_bounds__(256) void gemm_moe(
    const AT* __restrict__ A, int lda, int kp,
    const unsigned int* __restrict__ Wp, int npad,
    const float* __restrict__ bias,
    const float* __restrict__ omega,
    OT* __restrict__ C, int ldc, int nreal)
{
  // per buffer: 4 subtiles * 32 lanes * 12 dwords (8 data + 4 pad) = 1536 dwords
  __shared__ unsigned int lds_b[2 * 1536];
  const int t    = threadIdx.x;
  const int wv   = t >> 5;
  const int lane = t & 31;
  const int lm   = lane & 15;
  const int hl   = lane >> 4;
  const int tile_m = blockIdx.y * 128 + wv * 16;
  const int n0     = blockIdx.x * 64;
  const AT* arow = A + (size_t)(tile_m + lm) * lda;
  const int NK     = kp >> 5;
  const int kchunk = npad << 4;          // dwords per K-step in Wp (per expert)

  // staging decomposition: t -> (subtile sj, lane slane, dword-group srg)
  const int sj    = t >> 6;
  const int srem  = t & 63;
  const int slane = srem >> 1;
  const int srg   = (srem & 1) << 2;
  const int sdst  = sj * 384 + slane * 12 + srg;   // dword offset in one buffer
  const int sgsrc = t << 2;                        // dword offset in 4KB chunk
  unsigned int* sd0 = &lds_b[sdst];
  unsigned int* sd1 = &lds_b[1536 + sdst];

  f32x8 acct[4];
#pragma unroll
  for (int j = 0; j < 4; ++j) acct[j] = zero8();

  for (int e = 0; e < EXP; ++e) {
    f32x8 acc[4];
#pragma unroll
    for (int j = 0; j < 4; ++j) acc[j] = zero8();

    const unsigned int* We = Wp + (size_t)e * ((size_t)(kp >> 1) * (size_t)npad);
    const unsigned int* gs = We + (size_t)(n0 >> 4) * 256 + sgsrc;

    __syncthreads();                      // all prior reads of lds_b complete
    stage_b128(gs, sd0);                  // prologue: tile 0 -> buf0

    const int NK1 = NK - 1;
    for (int ks = 0; ks < NK1; ++ks) {    // steady state: branch-free body
      __syncthreads();                    // (A) reads of next buffer (iter ks-1) done
      gs += kchunk;
      stage_b128(gs, (ks & 1) ? sd0 : sd1);        // tile ks+1 -> other buffer
      u32x8 au = load_a_frag(arow, ks * 32, hl);   // independent of LDS
      wait_async1();                      // in-order: tile ks landed
      __syncthreads();                    // (B) tile ks visible block-wide
      mma_step(&lds_b[(ks & 1) * 1536], lane, au, acc);
    }
    {                                     // peeled last K-step (no stage)
      __syncthreads();
      u32x8 au = load_a_frag(arow, NK1 * 32, hl);
      wait_async0();
      __syncthreads();
      mma_step(&lds_b[(NK1 & 1) * 1536], lane, au, acc);
    }

    if (EXP > 1) {
      float wr[8];
#pragma unroll
      for (int r = 0; r < 8; ++r)
        wr[r] = omega[(size_t)(tile_m + hl * 8 + r) * 8 + e];
#pragma unroll
      for (int j = 0; j < 4; ++j) {
        float be = bias[e * npad + n0 + j * 16 + lm];
#pragma unroll
        for (int r = 0; r < 8; ++r)
          acct[j][r] += wr[r] * (acc[j][r] + be);
      }
    } else {
#pragma unroll
      for (int j = 0; j < 4; ++j) {
        float be = bias[n0 + j * 16 + lm];
#pragma unroll
        for (int r = 0; r < 8; ++r)
          acct[j][r] = acc[j][r] + be;
      }
    }
  }

#pragma unroll
  for (int j = 0; j < 4; ++j) {
    int n = n0 + j * 16 + lm;
    if (n < nreal) {
#pragma unroll
      for (int r = 0; r < 8; ++r) {
        float v = acct[j][r];
        if (ACT) v = elu1(v);
        stout(&C[(size_t)(tile_m + hl * 8 + r) * ldc + n], v);
      }
    }
  }
}

// ---------- weight packing: fp32 [E][K][N] -> fragment-swizzled bf16-pair dwords
// layout: [e][ks][jg][lane(32)][r(8)]; value = pair(K=ks*32+16*(r>>2)+8*(lane>>4)+2*(r&3),
//                                                  N=jg*16+(lane&15)); zero-padded.
__global__ void pack_w_sw(const float* __restrict__ W, unsigned int* __restrict__ Wp,
                          int K, int N, int kp, int npad, int total) {
  int idx = blockIdx.x * 256 + threadIdx.x;
  if (idx >= total) return;
  int perE = (kp >> 1) * npad;
  int e = idx / perE, rem = idx - e * perE;
  int ks   = rem / (npad << 4);
  int rem2 = rem - ks * (npad << 4);
  int jg   = rem2 >> 8;
  int rem3 = rem2 & 255;
  int ln   = rem3 >> 3;
  int r    = rem3 & 7;
  int k = ks * 32 + 16 * (r >> 2) + 8 * (ln >> 4) + 2 * (r & 3);
  int n = jg * 16 + (ln & 15);
  float lo = (k < K && n < N)     ? W[((size_t)e * K + k) * N + n]     : 0.f;
  float hi = (k + 1 < K && n < N) ? W[((size_t)e * K + k + 1) * N + n] : 0.f;
  Wp[idx] = pack2(lo, hi);
}

__global__ void pad_bias(const float* __restrict__ b, float* __restrict__ bp,
                         int N, int npad, int total) {
  int idx = blockIdx.x * 256 + threadIdx.x;
  if (idx >= total) return;
  int e = idx / npad, n = idx - e * npad;
  bp[idx] = (n < N) ? b[(size_t)e * N + n] : 0.f;
}

// ---------- h0 = [z | p_prev | (Ienc later) | zeros], bf16 [B, 832] ----------
__global__ void build_h0(const float* __restrict__ z, const float* __restrict__ p,
                         unsigned short* __restrict__ h0, int total) {
  int idx = blockIdx.x * 256 + threadIdx.x;
  if (idx >= total) return;
  int b = idx / 832, c = idx - b * 832;
  float v = 0.f;
  if (c < 32)       v = z[b * 32 + c];
  else if (c < 556) v = p[(size_t)b * 524 + (c - 32)];
  h0[idx] = f2bf(v);
}

// z into cols 0..31 of both [B,288] concat buffers
__global__ void init_hz(const float* __restrict__ z,
                        unsigned short* __restrict__ hz1,
                        unsigned short* __restrict__ hz2, int total) {
  int idx = blockIdx.x * 256 + threadIdx.x;
  if (idx >= total) return;
  int b = idx >> 5, c = idx & 31;
  unsigned short v = f2bf(z[idx]);
  hz1[b * 288 + c] = v;
  hz2[b * 288 + c] = v;
}

// softmax over 5 logits -> omega [B,8] (pad zeros)
__global__ void softmax5(const float* __restrict__ logits, float* __restrict__ omega, int B) {
  int b = blockIdx.x * 256 + threadIdx.x;
  if (b >= B) return;
  float v[5], mx = -3.0e38f;
#pragma unroll
  for (int j = 0; j < 5; ++j) { v[j] = logits[b * 8 + j]; mx = fmaxf(mx, v[j]); }
  float s = 0.f;
#pragma unroll
  for (int j = 0; j < 5; ++j) { v[j] = __expf(v[j] - mx); s += v[j]; }
  float inv = 1.f / s;
#pragma unroll
  for (int j = 0; j < 5; ++j) omega[b * 8 + j] = v[j] * inv;
  omega[b * 8 + 5] = 0.f; omega[b * 8 + 6] = 0.f; omega[b * 8 + 7] = 0.f;
}

// ---------- host ----------
extern "C" void kernel_launch(void* const* d_in, const int* in_sizes, int n_in,
                              void* d_out, int out_size, void* d_ws, size_t ws_size,
                              hipStream_t stream) {
  const int B = 16384;
  const float* z    = (const float*)d_in[0];
  const float* pprev= (const float*)d_in[1];
  const float* I    = (const float*)d_in[2];
  const float* gw1  = (const float*)d_in[3];  const float* gb1 = (const float*)d_in[4];
  const float* gw2  = (const float*)d_in[5];  const float* gb2 = (const float*)d_in[6];
  const float* gw3  = (const float*)d_in[7];  const float* gb3 = (const float*)d_in[8];
  const float* iw1  = (const float*)d_in[9];  const float* ib1 = (const float*)d_in[10];
  const float* iw2  = (const float*)d_in[11]; const float* ib2 = (const float*)d_in[12];
  const float* iw3  = (const float*)d_in[13]; const float* ib3 = (const float*)d_in[14];
  const float* wl1  = (const float*)d_in[15]; const float* bl1 = (const float*)d_in[16];
  const float* wl2  = (const float*)d_in[17]; const float* bl2 = (const float*)d_in[18];
  const float* wl3  = (const float*)d_in[19]; const float* bl3 = (const float*)d_in[20];

  char* ws = (char*)d_ws;
  size_t off = 0;
  auto alloc = [&](size_t bytes) -> char* {
    char* p = ws + off;
    off += (bytes + 255) & ~(size_t)255;
    return p;
  };

  unsigned short* h0  = (unsigned short*)alloc((size_t)B * 832 * 2);
  unsigned short* g1  = (unsigned short*)alloc((size_t)B * 256 * 2);
  unsigned short* g2  = (unsigned short*)alloc((size_t)B * 128 * 2);
  float*          lgt = (float*)alloc((size_t)B * 8 * 4);
  float*          omg = (float*)alloc((size_t)B * 8 * 4);
  unsigned short* i1b = (unsigned short*)alloc((size_t)B * 256 * 2);
  unsigned short* i2b = (unsigned short*)alloc((size_t)B * 256 * 2);
  unsigned short* hz1 = (unsigned short*)alloc((size_t)B * 288 * 2);
  unsigned short* hz2 = (unsigned short*)alloc((size_t)B * 288 * 2);
  unsigned int* wg1p = (unsigned int*)alloc((size_t)288 * 256 * 4);
  unsigned int* wg2p = (unsigned int*)alloc((size_t)128 * 128 * 4);
  unsigned int* wg3p = (unsigned int*)alloc((size_t)64 * 64 * 4);
  unsigned int* wi1p = (unsigned int*)alloc((size_t)1024 * 256 * 4);
  unsigned int* wi2p = (unsigned int*)alloc((size_t)128 * 256 * 4);
  unsigned int* wi3p = (unsigned int*)alloc((size_t)128 * 256 * 4);
  unsigned int* wl1p = (unsigned int*)alloc((size_t)5 * 416 * 256 * 4);
  unsigned int* wl2p = (unsigned int*)alloc((size_t)5 * 144 * 256 * 4);
  unsigned int* wl3p = (unsigned int*)alloc((size_t)5 * 144 * 576 * 4);
  float* bg1p = (float*)alloc(256 * 4);
  float* bg2p = (float*)alloc(128 * 4);
  float* bg3p = (float*)alloc(64 * 4);
  float* bi1p = (float*)alloc(256 * 4);
  float* bi2p = (float*)alloc(256 * 4);
  float* bi3p = (float*)alloc(256 * 4);
  float* bl1p = (float*)alloc(5 * 256 * 4);
  float* bl2p = (float*)alloc(5 * 256 * 4);
  float* bl3p = (float*)alloc(5 * 576 * 4);
  (void)ws_size; (void)in_sizes; (void)n_in; (void)out_size;

  dim3 blk(256);
  // inputs / concat buffers
  { int tot = B * 832; build_h0<<<(tot + 255) / 256, blk, 0, stream>>>(z, pprev, h0, tot); }
  { int tot = B * 32;  init_hz<<<(tot + 255) / 256, blk, 0, stream>>>(z, hz1, hz2, tot); }

  // weight packing (fragment-swizzled bf16 pairs, zero-padded K/N)
  auto packw = [&](const float* W, unsigned int* Wp, int K, int N, int kp, int npad, int E) {
    int tot = E * (kp / 2) * npad;
    pack_w_sw<<<(tot + 255) / 256, blk, 0, stream>>>(W, Wp, K, N, kp, npad, tot);
  };
  auto padb = [&](const float* b, float* bp, int N, int npad, int E) {
    int tot = E * npad;
    pad_bias<<<(tot + 255) / 256, blk, 0, stream>>>(b, bp, N, npad, tot);
  };
  packw(gw1, wg1p, 556, 256, 576, 256, 1);   padb(gb1, bg1p, 256, 256, 1);
  packw(gw2, wg2p, 256, 128, 256, 128, 1);   padb(gb2, bg2p, 128, 128, 1);
  packw(gw3, wg3p, 128, 5, 128, 64, 1);      padb(gb3, bg3p, 5, 64, 1);
  packw(iw1, wi1p, 2048, 256, 2048, 256, 1); padb(ib1, bi1p, 256, 256, 1);
  packw(iw2, wi2p, 256, 256, 256, 256, 1);   padb(ib2, bi2p, 256, 256, 1);
  packw(iw3, wi3p, 256, 256, 256, 256, 1);   padb(ib3, bi3p, 256, 256, 1);
  packw(wl1, wl1p, 812, 256, 832, 256, 5);   padb(bl1, bl1p, 256, 256, 5);
  packw(wl2, wl2p, 288, 256, 288, 256, 5);   padb(bl2, bl2p, 256, 256, 5);
  packw(wl3, wl3p, 288, 524, 288, 576, 5);   padb(bl3, bl3p, 524, 576, 5);

  const int GY = B / 128;  // 128 M-blocks

  // gating: h0[:, :556] -> 256 -> 128 -> 5 -> softmax
  gemm_moe<1, true,  unsigned short, unsigned short><<<dim3(4, GY), blk, 0, stream>>>(
      h0, 832, 576, wg1p, 256, bg1p, nullptr, g1, 256, 256);
  gemm_moe<1, true,  unsigned short, unsigned short><<<dim3(2, GY), blk, 0, stream>>>(
      g1, 256, 256, wg2p, 128, bg2p, nullptr, g2, 128, 128);
  gemm_moe<1, false, unsigned short, float><<<dim3(1, GY), blk, 0, stream>>>(
      g2, 128, 128, wg3p, 64, bg3p, nullptr, lgt, 8, 5);
  softmax5<<<(B + 255) / 256, blk, 0, stream>>>(lgt, omg, B);

  // INet: I(fp32) -> 256 -> 256 -> Ienc, written into h0 cols 556..811
  gemm_moe<1, true, float, unsigned short><<<dim3(4, GY), blk, 0, stream>>>(
      I, 2048, 2048, wi1p, 256, bi1p, nullptr, i1b, 256, 256);
  gemm_moe<1, true, unsigned short, unsigned short><<<dim3(4, GY), blk, 0, stream>>>(
      i1b, 256, 256, wi2p, 256, bi2p, nullptr, i2b, 256, 256);
  gemm_moe<1, true, unsigned short, unsigned short><<<dim3(4, GY), blk, 0, stream>>>(
      i2b, 256, 256, wi3p, 256, bi3p, nullptr, h0 + 556, 832, 256);

  // MoE layers (5 experts, omega-blended)
  gemm_moe<5, true, unsigned short, unsigned short><<<dim3(4, GY), blk, 0, stream>>>(
      h0, 832, 832, wl1p, 256, bl1p, omg, hz1 + 32, 288, 256);
  gemm_moe<5, true, unsigned short, unsigned short><<<dim3(4, GY), blk, 0, stream>>>(
      hz1, 288, 288, wl2p, 256, bl2p, omg, hz2 + 32, 288, 256);
  gemm_moe<5, false, unsigned short, float><<<dim3(9, GY), blk, 0, stream>>>(
      hz2, 288, 288, wl3p, 576, bl3p, omg, (float*)d_out, 524, 524);
}
